// SingleHeadedAttention_1340029796391
// MI455X (gfx1250) — matrix operand order repaired and checked by reading, not compiled, not run on hardware
//
#include <hip/hip_runtime.h>
#include <hip/hip_bf16.h>

typedef __attribute__((ext_vector_type(16))) _Float16 v16h;
typedef __attribute__((ext_vector_type(8)))  _Float16 v8h;
typedef __attribute__((ext_vector_type(8)))  float    v8f;

constexpr int kDModel = 1024;
constexpr int kDK     = 128;   // = d_v
constexpr int kSeq    = 4096;

// LDS strides padded to odd multiples of 16B -> conflict-free b128 gathers
constexpr int kKRowB = 272;    // 32 keys x 256B rows, padded (+16)
constexpr int kVRowB = 80;     // 128 dims x 64B rows, padded (+16)
constexpr int kPRowB = 80;     // 16 rows x 64B, padded (+16)

static __device__ __forceinline__ v8f wmma_f16(v16h a, v16h b, v8f c) {
  // D = A(16x32 f16) * B(32x16 f16) + C(16x16 f32)
  return __builtin_amdgcn_wmma_f32_16x16x32_f16(
      /*neg_a=*/false, a, /*neg_b=*/false, b,
      /*c_mod=*/(short)0, c, /*reuse_a=*/false, /*reuse_b=*/false);
}

// Async DMA: 16 bytes per lane, global -> LDS. Tracked by ASYNCcnt.
static __device__ __forceinline__ void async_ld16(const void* g, unsigned lds_off) {
  unsigned long long ga = (unsigned long long)(uintptr_t)g;
  asm volatile("global_load_async_to_lds_b128 %0, %1, off"
               :: "v"(lds_off), "v"(ga) : "memory");
}
static __device__ __forceinline__ void wait_async0() {
  asm volatile("s_wait_asynccnt 0" ::: "memory");
}

// ---------------------------------------------------------------------------
// Prep: transpose + convert [1024][128] f32 weight into [128][1024] f16 so
// WMMA B-fragments become contiguous 16B loads per lane.
// ---------------------------------------------------------------------------
__global__ __launch_bounds__(256) void wcvt_kernel(const float* __restrict__ w,
                                                   _Float16* __restrict__ wt) {
  int idx = blockIdx.x * 256 + threadIdx.x;     // over 1024*128
  if (idx < kDModel * kDK) {
    int k = idx >> 7;
    int n = idx & (kDK - 1);
    wt[n * kDModel + k] = (_Float16)w[idx];
  }
}

// ---------------------------------------------------------------------------
// Projection (templated on which matrix): one wave computes a 16x128 tile.
// MAT 0/1 (Q/K): row-major f16 [16384][128].  MAT 2 (V): f16 [b][128][4096].
// ---------------------------------------------------------------------------
template <int MAT>
__global__ __launch_bounds__(256) void proj_kernel(
    const float* __restrict__ x,        // [16384][1024]
    const _Float16* __restrict__ wm,    // [128][1024] transposed f16 weight
    const float* __restrict__ bias,     // [128]
    _Float16* __restrict__ dst) {
  const int lane = threadIdx.x & 31;
  const int wave = threadIdx.x >> 5;
  const int tile = blockIdx.x * 8 + wave;   // 0..1023
  const int row0 = tile * 16;
  const int nlo  = lane & 15;
  const int hi   = lane >> 4;

  const float* xrow = x + (size_t)(row0 + nlo) * kDModel;

  v8f acc[8];
#pragma unroll
  for (int nt = 0; nt < 8; ++nt) {
    float bvl = bias[nt * 16 + nlo];          // bias depends on N only
#pragma unroll
    for (int j = 0; j < 8; ++j) acc[nt][j] = bvl;
  }

  for (int kb = 0; kb < kDModel; kb += 32) {
    // A fragment: lane holds (m = nlo, k = 16*hi + i)
    float af[16];
    *(float4*)(af + 0)  = *(const float4*)(xrow + kb + 16 * hi);
    *(float4*)(af + 4)  = *(const float4*)(xrow + kb + 16 * hi + 4);
    *(float4*)(af + 8)  = *(const float4*)(xrow + kb + 16 * hi + 8);
    *(float4*)(af + 12) = *(const float4*)(xrow + kb + 16 * hi + 12);
    v16h a;
#pragma unroll
    for (int i = 0; i < 16; ++i) a[i] = (_Float16)af[i];

    // Preload all 8 B fragments, then 8 back-to-back WMMAs.
    v16h bf[8];
#pragma unroll
    for (int nt = 0; nt < 8; ++nt) {
      const _Float16* wrow = wm + (size_t)(nt * 16 + nlo) * kDModel + kb + 16 * hi;
      v8h b0 = *(const v8h*)(wrow);
      v8h b1 = *(const v8h*)(wrow + 8);
#pragma unroll
      for (int i = 0; i < 8; ++i) { bf[nt][i] = b0[i]; bf[nt][i + 8] = b1[i]; }
    }
#pragma unroll
    for (int nt = 0; nt < 8; ++nt) acc[nt] = wmma_f16(a, bf[nt], acc[nt]);
  }

  // C layout: element (M = j + 8*hi, N = nt*16 + nlo)
#pragma unroll
  for (int nt = 0; nt < 8; ++nt) {
#pragma unroll
    for (int j = 0; j < 8; ++j) {
      int row = row0 + j + 8 * hi;
      int col = nt * 16 + nlo;
      _Float16 h = (_Float16)acc[nt][j];
      if constexpr (MAT == 2) {
        int b = row >> 12;                 // /4096
        int s = row & (kSeq - 1);
        dst[((size_t)b * kDK + col) * kSeq + s] = h;   // V transposed
      } else {
        dst[(size_t)row * kDK + col] = h;              // Q / K row-major
      }
    }
  }
}

// ---------------------------------------------------------------------------
// Flash attention. 8 waves/block share one batch; K/V 32-key blocks are
// DMA'd to LDS once per block (async, double-buffered, padded layouts),
// then each wave gathers WMMA fragments from LDS. 16 WMMAs/key-block/wave.
// ---------------------------------------------------------------------------
__global__ __launch_bounds__(256) void attn_kernel(
    const _Float16* __restrict__ Q, const _Float16* __restrict__ K,
    const _Float16* __restrict__ Vt, float* __restrict__ out) {
  __shared__ __align__(16) char kbuf[2][32 * kKRowB];    // 2 x 8704B
  __shared__ __align__(16) char vbuf[2][kDK * kVRowB];   // 2 x 10240B
  __shared__ __align__(16) char pbuf[8][16 * kPRowB];    // 8 x 1280B

  const int lane = threadIdx.x & 31;
  const int wave = threadIdx.x >> 5;
  const int tile = blockIdx.x * 8 + wave;   // 0..1023; all waves: same batch
  const int b    = tile >> 8;               // 256 q-tiles per batch
  const int q0   = (tile & 255) * 16;
  const int nlo  = lane & 15;
  const int hi   = lane >> 4;

  const _Float16* Kb = K + (size_t)b * kSeq * kDK;
  const _Float16* Vb = Vt + (size_t)b * kDK * kSeq;

  // Cooperative async fill of one 32-key K block + V block (16KB payload,
  // 4 x b128 per thread across 256 threads), into padded LDS layouts.
  auto issue_block = [&](int kb, int cur) {
    const int tid = threadIdx.x;
    const unsigned kbase = (unsigned)(uintptr_t)&kbuf[cur][0];
    const unsigned vbase = (unsigned)(uintptr_t)&vbuf[cur][0];
#pragma unroll
    for (int r = 0; r < 2; ++r) {
      int o = (tid + 256 * r) * 16;                    // byte offset in 8KB payload
      // K block: contiguous 8KB in global, padded rows (272B) in LDS
      int krow = o >> 8, kcol = o & 255;
      async_ld16((const char*)Kb + (size_t)kb * (kDK * 2) + o,
                 kbase + krow * kKRowB + kcol);
      // V block: per dim, 64B slice [kb, kb+32), padded rows (80B) in LDS
      int dim = o >> 6, off = o & 63;
      async_ld16((const char*)Vb + ((size_t)dim * kSeq + kb) * 2 + off,
                 vbase + dim * kVRowB + off);
    }
  };

  // Q fragments (A layout), K-dim = 128 -> 4 chunks of 32
  const _Float16* qrow = Q + ((size_t)b * kSeq + q0 + nlo) * kDK;
  v16h qf[4];
#pragma unroll
  for (int c = 0; c < 4; ++c) {
    v8h lo8 = *(const v8h*)(qrow + c * 32 + 8 * hi);
    v8h hi8 = *(const v8h*)(qrow + c * 32 + 16 + 8 * hi);
#pragma unroll
    for (int i = 0; i < 8; ++i) { qf[c][i] = lo8[i]; qf[c][i + 8] = hi8[i]; }
  }

  float mrow[8], lrow[8];
#pragma unroll
  for (int j = 0; j < 8; ++j) { mrow[j] = -1e30f; lrow[j] = 0.0f; }
  v8f o[8];
#pragma unroll
  for (int nt = 0; nt < 8; ++nt)
#pragma unroll
    for (int j = 0; j < 8; ++j) o[nt][j] = 0.0f;

  char* pw = &pbuf[wave][0];
  const float scale = 1.0f / 128.0f;   // reference divides by d_k (not sqrt)

  issue_block(0, 0);
  int cur = 0;
  for (int kb = 0; kb < kSeq; kb += 32) {
    wait_async0();        // my share of buf[cur] landed
    __syncthreads();      // everyone's share landed; prev compute done
    if (kb + 32 < kSeq) issue_block(kb + 32, cur ^ 1);   // overlap next DMA

    const char* kl = &kbuf[cur][0];
    const char* vl = &vbuf[cur][0];

    // ---- scores S = Q @ K^T for 32 keys (two 16-col C tiles) ----
    v8f s0, s1;
#pragma unroll
    for (int j = 0; j < 8; ++j) { s0[j] = 0.0f; s1[j] = 0.0f; }
#pragma unroll
    for (int c = 0; c < 4; ++c) {
      // B-fragment rows: key = g*16 + nlo, 8 halves at c*64+16*hi and 8 at c*64+32+16*hi
      const char* k0 = kl + (size_t)nlo * kKRowB + c * 64 + 16 * hi;
      const char* k1 = kl + (size_t)(16 + nlo) * kKRowB + c * 64 + 16 * hi;
      v8h t0 = *(const v8h*)k0, t1 = *(const v8h*)(k0 + 32);
      v8h u0 = *(const v8h*)k1, u1 = *(const v8h*)(k1 + 32);
      v16h bf0, bf1;
#pragma unroll
      for (int i = 0; i < 8; ++i) {
        bf0[i] = t0[i]; bf0[i + 8] = t1[i];
        bf1[i] = u0[i]; bf1[i + 8] = u1[i];
      }
      s0 = wmma_f16(qf[c], bf0, s0);
      s1 = wmma_f16(qf[c], bf1, s1);
    }

    // ---- online softmax (row M = j + 8*hi lives in this lane's group) ----
    float corr[8];
#pragma unroll
    for (int j = 0; j < 8; ++j) {
      float a0 = s0[j] * scale, a1 = s1[j] * scale;
      float mx = fmaxf(a0, a1);
#pragma unroll
      for (int off = 1; off < 16; off <<= 1)
        mx = fmaxf(mx, __shfl_xor(mx, off, 32));
      float mnew = fmaxf(mrow[j], mx);
      corr[j] = __expf(mrow[j] - mnew);
      float p0 = __expf(a0 - mnew);
      float p1 = __expf(a1 - mnew);
      float rs = p0 + p1;
#pragma unroll
      for (int off = 1; off < 16; off <<= 1)
        rs += __shfl_xor(rs, off, 32);
      lrow[j] = lrow[j] * corr[j] + rs;
      mrow[j] = mnew;
      // stage P (f16): row (j+8*hi), cols nlo and 16+nlo (padded 80B rows)
      *(_Float16*)(pw + (j + 8 * hi) * kPRowB + 2 * nlo)      = (_Float16)p0;
      *(_Float16*)(pw + (j + 8 * hi) * kPRowB + 32 + 2 * nlo) = (_Float16)p1;
    }
#pragma unroll
    for (int nt = 0; nt < 8; ++nt)
#pragma unroll
      for (int j = 0; j < 8; ++j) o[nt][j] *= corr[j];

    // LDS store -> gather hazard within this wave (CDNA5 split counter)
    asm volatile("s_wait_dscnt 0" ::: "memory");

    // gather P as A fragment (m = nlo, k = (i>>3)*16 + (i&7) + 8*hi)
    v16h pf;
    {
      v8h pl = *(const v8h*)(pw + nlo * kPRowB + 16 * hi);
      v8h ph = *(const v8h*)(pw + nlo * kPRowB + 32 + 16 * hi);
#pragma unroll
      for (int i = 0; i < 8; ++i) { pf[i] = pl[i]; pf[i + 8] = ph[i]; }
    }

    // ---- O += P @ V (V staged dims x keys, padded rows) ----
#pragma unroll
    for (int nt = 0; nt < 8; ++nt) {
      const char* vp = vl + (size_t)(nt * 16 + nlo) * kVRowB + 16 * hi;
      v8h v0 = *(const v8h*)vp, v1 = *(const v8h*)(vp + 32);
      v16h vf;
#pragma unroll
      for (int i = 0; i < 8; ++i) { vf[i] = v0[i]; vf[i + 8] = v1[i]; }
      o[nt] = wmma_f16(pf, vf, o[nt]);
    }
    cur ^= 1;
  }

  // ---- normalize + write out (f32) ----
  float* orow = out + ((size_t)b * kSeq + q0) * kDK;
#pragma unroll
  for (int j = 0; j < 8; ++j) {
    float inv = 1.0f / lrow[j];
#pragma unroll
    for (int nt = 0; nt < 8; ++nt)
      orow[(size_t)(j + 8 * hi) * kDK + nt * 16 + nlo] = o[nt][j] * inv;
  }
}

// ---------------------------------------------------------------------------
extern "C" void kernel_launch(void* const* d_in, const int* in_sizes, int n_in,
                              void* d_out, int out_size, void* d_ws, size_t ws_size,
                              hipStream_t stream) {
  const float* x  = (const float*)d_in[0];
  const float* wq = (const float*)d_in[1];
  const float* bq = (const float*)d_in[2];
  const float* wk = (const float*)d_in[3];
  const float* bk = (const float*)d_in[4];
  const float* wv = (const float*)d_in[5];
  const float* bv = (const float*)d_in[6];
  float* out = (float*)d_out;

  // workspace layout (f16):
  //   wt  : 3 * 128 * 1024  =   786,432 B
  //   Q   : 16384 * 128     = 4,194,304 B
  //   K   : 16384 * 128     = 4,194,304 B
  //   Vt  : 4 * 128 * 4096  = 4,194,304 B
  char* ws = (char*)d_ws;
  _Float16* wt = (_Float16*)ws;
  _Float16* Q  = (_Float16*)(ws + (size_t)786432);
  _Float16* K  = (_Float16*)(ws + (size_t)786432 + 4194304);
  _Float16* Vt = (_Float16*)(ws + (size_t)786432 + 2 * (size_t)4194304);

  _Float16* wtq = wt + 0 * (size_t)kDK * kDModel;
  _Float16* wtk = wt + 1 * (size_t)kDK * kDModel;
  _Float16* wtv = wt + 2 * (size_t)kDK * kDModel;

  // 1) weight transpose+convert (3 x 131072 elements)
  wcvt_kernel<<<512, 256, 0, stream>>>(wq, wtq);
  wcvt_kernel<<<512, 256, 0, stream>>>(wk, wtk);
  wcvt_kernel<<<512, 256, 0, stream>>>(wv, wtv);

  // 2) projections: 1024 wave-tiles of 16 rows, 8 waves/block
  proj_kernel<0><<<128, 256, 0, stream>>>(x, wtq, bq, Q);
  proj_kernel<1><<<128, 256, 0, stream>>>(x, wtk, bk, K);
  proj_kernel<2><<<128, 256, 0, stream>>>(x, wtv, bv, Vt);

  // 3) flash attention: 1024 wave-tiles (4 batches x 256 q-tiles), 8 waves/block
  attn_kernel<<<128, 256, 0, stream>>>(Q, K, Vt, out);
}
